// GCN_mamba_Net_16149077033110
// MI455X (gfx1250) — matrix operand order repaired and checked by reading, not compiled, not run on hardware
//
#include <hip/hip_runtime.h>
#include <stdint.h>

// ---------------- problem constants ----------------
#define N_NODES 2048
#define F_IN    1024
#define DMD     256
#define DI      256
#define DS_Gc   8
#define DTR_Gc  16
#define DS_Bc   16
#define DTR_Bc  16
#define L_GCN   8
#define KCONV   4
#define ALPHAc  0.1f
#define GWc     0.8f
#define BLKA    0.05f
#define BN_EPS  1e-5f
#define NM      (N_NODES*DMD)   // 524288

typedef __attribute__((ext_vector_type(16))) __bf16 v16bf;
typedef __attribute__((ext_vector_type(8)))  __bf16 v8bf;
typedef __attribute__((ext_vector_type(8)))  float  v8f;

// ---------------- device helpers ----------------
__device__ __forceinline__ __bf16 f2bf(float f) {
  unsigned u = __builtin_bit_cast(unsigned, f);
  unsigned short s = (unsigned short)(u >> 16);       // truncate mantissa -> bf16
  return __builtin_bit_cast(__bf16, s);
}
__device__ __forceinline__ float siluf(float x)     { return x / (1.f + __expf(-x)); }
__device__ __forceinline__ float softplusf(float x) { return (x > 20.f) ? x : log1pf(__expf(x)); }

// ---------------- WMMA GEMM: C[M,N] = A[M,K] * BT[N,K]^T (+C if beta) ----------------
// A, BT stored bf16. M mult of 16, N mult of 32, K mult of 32.
// One wave computes a 16x32 tile (one A fragment, two B fragments, two accumulators).
// Inner loop: 6x b128 loads + 2x v_wmma, no conversion VALU.
__global__ __launch_bounds__(256) void k_gemm(const __bf16* __restrict__ A,
                                              const __bf16* __restrict__ BT,
                                              float* __restrict__ C,
                                              int M, int N, int K, int beta) {
  const int lane   = threadIdx.x & 31;
  const int wave   = threadIdx.x >> 5;
  const int tilesN = N >> 5;                    // 32-wide N tiles
  const int tile   = blockIdx.x * 8 + wave;
  if (tile >= (M >> 4) * tilesN) return;        // wave-uniform: EXEC stays all-1s
  const int tm   = (tile / tilesN) << 4;
  const int tn   = (tile % tilesN) << 5;
  const int half = lane >> 4;
  const int lidx = lane & 15;

  const __bf16* Arow  = A  + (size_t)(tm + lidx) * K;
  const __bf16* Brow0 = BT + (size_t)(tn + lidx) * K;
  const __bf16* Brow1 = BT + (size_t)(tn + 16 + lidx) * K;

  v8f acc0, acc1;
  if (beta) {
#pragma unroll
    for (int r = 0; r < 8; ++r) {
      acc0[r] = C[(size_t)(tm + r + 8 * half) * N + tn + lidx];
      acc1[r] = C[(size_t)(tm + r + 8 * half) * N + tn + 16 + lidx];
    }
  } else {
#pragma unroll
    for (int r = 0; r < 8; ++r) { acc0[r] = 0.f; acc1[r] = 0.f; }
  }

#pragma unroll 2
  for (int kk = 0; kk < K; kk += 32) {
    v8bf alo = *(const v8bf*)(Arow + kk + half * 16);       // A j=0..7
    v8bf ahi = *(const v8bf*)(Arow + kk + half * 16 + 8);   // A j=8..15
    v16bf a  = __builtin_shufflevector(alo, ahi,
                 0, 1, 2, 3, 4, 5, 6, 7, 8, 9, 10, 11, 12, 13, 14, 15);
    v16bf b0 = *(const v16bf*)(Brow0 + kk + half * 16);     // B cols tn..tn+15
    v16bf b1 = *(const v16bf*)(Brow1 + kk + half * 16);     // B cols tn+16..tn+31
    __builtin_prefetch(Arow + kk + 64, 0, 1);
    __builtin_prefetch(Brow0 + kk + 64, 0, 1);
    acc0 = __builtin_amdgcn_wmma_f32_16x16x32_bf16(false, a, false, b0,
                                                   (short)0, acc0, false, false);
    acc1 = __builtin_amdgcn_wmma_f32_16x16x32_bf16(false, a, false, b1,
                                                   (short)0, acc1, false, false);
  }
#pragma unroll
  for (int r = 0; r < 8; ++r) {
    C[(size_t)(tm + r + 8 * half) * N + tn + lidx]      = acc0[r];
    C[(size_t)(tm + r + 8 * half) * N + tn + 16 + lidx] = acc1[r];
  }
}

// ---------------- conversion / transform kernels ----------------
// f32 -> bf16 elementwise
__global__ __launch_bounds__(256) void k_cvt_bf(__bf16* __restrict__ dst,
    const float* __restrict__ src, int n) {
  int idx = blockIdx.x * 256 + threadIdx.x;
  if (idx < n) dst[idx] = f2bf(src[idx]);
}

// dst[Nd,Kd](bf16) = transpose(src[K,N] f32), zero padded beyond K/N
__global__ __launch_bounds__(256) void k_transpose_bf(__bf16* __restrict__ dst,
    const float* __restrict__ src, int K, int N, int Kd, int Nd) {
  int idx = blockIdx.x * 256 + threadIdx.x;
  if (idx >= Nd * Kd) return;
  int n = idx / Kd, k = idx % Kd;
  dst[idx] = (n < N && k < K) ? f2bf(src[(size_t)k * N + n]) : f2bf(0.f);
}

// dst[R,Cd](bf16) = src[R,Cs](f32) zero-padded in columns
__global__ __launch_bounds__(256) void k_padcopy_bf(__bf16* __restrict__ dst,
    const float* __restrict__ src, int R, int Cs, int Cd) {
  int idx = blockIdx.x * 256 + threadIdx.x;
  if (idx >= R * Cd) return;
  int r = idx / Cd, c2 = idx % Cd;
  dst[idx] = (c2 < Cs) ? f2bf(src[(size_t)r * Cs + c2]) : f2bf(0.f);
}

__global__ __launch_bounds__(256) void k_neg_exp(float* __restrict__ dst,
    const float* __restrict__ src, int n) {
  int idx = blockIdx.x * 256 + threadIdx.x;
  if (idx < n) dst[idx] = -__expf(src[idx]);
}

// causal depthwise conv (K=4) + SiLU; writes f32 (scan input) and bf16 (GEMM A)
__global__ __launch_bounds__(256) void k_conv_silu(const float* __restrict__ xz,
    const float* __restrict__ cw, const float* __restrict__ cb,
    float* __restrict__ xc, __bf16* __restrict__ xcb, int dir) {
  int idx = blockIdx.x * 256 + threadIdx.x;
  if (idx >= N_NODES * DI) return;
  int l = idx / DI, d = idx % DI;
  float s = cb[d];
#pragma unroll
  for (int k = 0; k < KCONV; ++k) {
    int i = l - (KCONV - 1) + k;
    if (i < 0) continue;
    int p = dir ? (N_NODES - 1 - i) : i;
    s += cw[d * KCONV + k] * xz[(size_t)p * (2 * DI) + d];
  }
  float v = siluf(s);
  xc[idx]  = v;
  xcb[idx] = f2bf(v);
}

// global path: dt32(bf16)[l, 0..15] = x_dbl[l, 0..15], rest zero
__global__ __launch_bounds__(256) void k_split_g(__bf16* __restrict__ dt32,
    const float* __restrict__ xdbl) {
  int idx = blockIdx.x * 256 + threadIdx.x;
  if (idx >= N_NODES * 32) return;
  int c2 = idx & 31;
  dt32[idx] = (c2 < DTR_Gc) ? f2bf(xdbl[idx]) : f2bf(0.f);
}

// softplus with optional per-column bias
__global__ __launch_bounds__(256) void k_softplus(float* __restrict__ dst,
    const float* __restrict__ src, const float* __restrict__ bias,
    int n, int cols, int hasbias) {
  int idx = blockIdx.x * 256 + threadIdx.x;
  if (idx >= n) return;
  float v = src[idx];
  if (hasbias) v += bias[idx % cols];
  dst[idx] = softplusf(v);
}

// global selective scan: 256 threads (one per channel), 2048 sequential steps,
// fused y*silu(z), written back at original positions as bf16 (out-proj GEMM A)
__global__ __launch_bounds__(256) void k_scan_global(const float* __restrict__ dlt,
    const float* __restrict__ u, const float* __restrict__ xdbl,
    const float* __restrict__ gA, const float* __restrict__ gD,
    const float* __restrict__ xz, __bf16* __restrict__ yzb, int dir) {
  int d = threadIdx.x;
  float h[DS_Gc], Av[DS_Gc];
#pragma unroll
  for (int n = 0; n < DS_Gc; ++n) { h[n] = 0.f; Av[n] = gA[d * DS_Gc + n]; }
  float Dd = gD[d];
  for (int l = 0; l < N_NODES; ++l) {
    float dl = dlt[l * DI + d];
    float uv = u[l * DI + d];
    float du = dl * uv;
    const float* xb = xdbl + l * 32;
    float y = 0.f;
#pragma unroll
    for (int n = 0; n < DS_Gc; ++n) {
      h[n] = __expf(dl * Av[n]) * h[n] + du * xb[16 + n];
      y += h[n] * xb[24 + n];
    }
    y += uv * Dd;
    int p = dir ? (N_NODES - 1 - l) : l;
    float z = xz[(size_t)p * (2 * DI) + DI + d];
    yzb[(size_t)p * DI + d] = f2bf(y * siluf(z));
  }
}

// gafin = relu(ga)*(1-ALPHA) + ALPHA*xin
__global__ __launch_bounds__(256) void k_gafinal(float* __restrict__ dst,
    const float* __restrict__ ga, const float* __restrict__ xin, int n) {
  int idx = blockIdx.x * 256 + threadIdx.x;
  if (idx < n) dst[idx] = fmaxf(ga[idx], 0.f) * (1.f - ALPHAc) + ALPHAc * xin[idx];
}

// relu(batchnorm(x)); writes f32 and bf16 copies
__global__ __launch_bounds__(256) void k_bn_relu(float* __restrict__ dst,
    __bf16* __restrict__ dstb, const float* __restrict__ src,
    const float* __restrict__ g, const float* __restrict__ b,
    const float* __restrict__ m, const float* __restrict__ v, int n, int cols) {
  int idx = blockIdx.x * 256 + threadIdx.x;
  if (idx >= n) return;
  int d = idx % cols;
  float o = (src[idx] - m[d]) * rsqrtf(v[d] + BN_EPS) * g[d] + b[d];
  o = fmaxf(o, 0.f);
  dst[idx]  = o;
  dstb[idx] = f2bf(o);
}

// dst = (1-a)*src + a*base; writes f32 and bf16 copies
__global__ __launch_bounds__(256) void k_blend(float* __restrict__ dst,
    __bf16* __restrict__ dstb, const float* __restrict__ src,
    const float* __restrict__ base, int n, float a) {
  int idx = blockIdx.x * 256 + threadIdx.x;
  if (idx >= n) return;
  float o = (1.f - a) * src[idx] + a * base[idx];
  dst[idx]  = o;
  dstb[idx] = f2bf(o);
}

// relu + split of xs @ xp_w result (raw stride 64): dt(bf16, padded to 32), B, C (f32)
__global__ __launch_bounds__(256) void k_split_b(__bf16* __restrict__ dt32,
    float* __restrict__ Bo, float* __restrict__ Co, const float* __restrict__ raw) {
  int idx = blockIdx.x * 256 + threadIdx.x;
  if (idx >= N_NODES * 64) return;
  int r = idx >> 6, c2 = idx & 63;
  const float* rr = raw + (size_t)r * 64;
  if (c2 < 32)
    dt32[(size_t)r * 32 + c2] = (c2 < DTR_Bc) ? f2bf(fmaxf(rr[c2], 0.f)) : f2bf(0.f);
  else if (c2 < 48)
    Bo[(size_t)r * DS_Bc + (c2 - 32)] = fmaxf(rr[c2 - 16], 0.f);
  else
    Co[(size_t)r * DS_Bc + (c2 - 48)] = fmaxf(rr[c2 - 16], 0.f);
}

// block selective scan: fully parallel over (node, channel); L=8 steps; last y only (bf16)
__global__ __launch_bounds__(256) void k_scan_block(__bf16* __restrict__ yb,
    const float* __restrict__ ubuf, const float* __restrict__ dbuf,
    const float* __restrict__ Bbuf, const float* __restrict__ C7,
    const float* __restrict__ bA, const float* __restrict__ bD) {
  int idx = blockIdx.x * 256 + threadIdx.x;
  if (idx >= NM) return;
  int b = idx / DMD, d = idx % DMD;
  float h[DS_Bc], Av[DS_Bc];
#pragma unroll
  for (int n = 0; n < DS_Bc; ++n) { h[n] = 0.f; Av[n] = bA[d * DS_Bc + n]; }
  float uv = 0.f;
  for (int l = 0; l < L_GCN; ++l) {
    float dl = dbuf[(size_t)l * NM + idx];
    uv = ubuf[(size_t)l * NM + idx];
    float du = dl * uv;
    const float* Bl = Bbuf + (size_t)l * N_NODES * DS_Bc + (size_t)b * DS_Bc;
#pragma unroll
    for (int n = 0; n < DS_Bc; ++n)
      h[n] = __expf(dl * Av[n]) * h[n] + du * Bl[n];
  }
  const float* Cr = C7 + (size_t)b * DS_Bc;
  float yy = 0.f;
#pragma unroll
  for (int n = 0; n < DS_Bc; ++n) yy += h[n] * Cr[n];
  yb[idx] = f2bf(yy + uv * bD[d]);
}

// final: local = lpre + xin; out = relu(bn2(GW*local + (1-GW)*gafin));
// writes the 4 concatenated outputs (output, x_input, ga_final, local_output)
__global__ __launch_bounds__(256) void k_final(float* __restrict__ out,
    const float* __restrict__ lpre, const float* __restrict__ xin,
    const float* __restrict__ gafin, const float* __restrict__ g2,
    const float* __restrict__ b2, const float* __restrict__ m2,
    const float* __restrict__ v2) {
  int idx = blockIdx.x * 256 + threadIdx.x;
  if (idx >= NM) return;
  int d = idx % DMD;
  float lo = lpre[idx] + xin[idx];
  float o = lo * GWc + gafin[idx] * (1.f - GWc);
  o = (o - m2[d]) * rsqrtf(v2[d] + BN_EPS) * g2[d] + b2[d];
  out[idx]          = fmaxf(o, 0.f);
  out[NM + idx]     = xin[idx];
  out[2 * NM + idx] = gafin[idx];
  out[3 * NM + idx] = lo;
}

// ---------------- host launcher ----------------
static inline dim3 ew(size_t n) { return dim3((unsigned)((n + 255) / 256)); }

extern "C" void kernel_launch(void* const* d_in, const int* in_sizes, int n_in,
                              void* d_out, int out_size, void* d_ws, size_t ws_size,
                              hipStream_t stream) {
  (void)in_sizes; (void)n_in; (void)out_size; (void)ws_size;
  const float* x        = (const float*)d_in[0];
  const float* adj      = (const float*)d_in[1];
  const float* lin1_w   = (const float*)d_in[2];
  const float* g_in_w   = (const float*)d_in[3];
  const float* g_conv_w = (const float*)d_in[4];
  const float* g_conv_b = (const float*)d_in[5];
  const float* g_xproj  = (const float*)d_in[6];
  const float* g_dtproj = (const float*)d_in[7];
  const float* g_dtb    = (const float*)d_in[8];
  const float* g_A_log  = (const float*)d_in[9];
  const float* g_D      = (const float*)d_in[10];
  const float* g_out_w  = (const float*)d_in[11];
  const float* bn1_g    = (const float*)d_in[12];
  const float* bn1_b    = (const float*)d_in[13];
  const float* bn1_m    = (const float*)d_in[14];
  const float* bn1_v    = (const float*)d_in[15];
  const float* b_xproj  = (const float*)d_in[16];
  const float* b_dtproj = (const float*)d_in[17];
  const float* b_A_log  = (const float*)d_in[18];
  const float* b_D      = (const float*)d_in[19];
  const float* b_out_w  = (const float*)d_in[20];
  const float* bn2_g    = (const float*)d_in[21];
  const float* bn2_b    = (const float*)d_in[22];
  const float* bn2_m    = (const float*)d_in[23];
  const float* bn2_v    = (const float*)d_in[24];

  uint8_t* base = (uint8_t*)d_ws;
  size_t cur = 0;
  auto allocB = [&](size_t bytes) {
    void* p = base + cur;
    cur = (cur + bytes + 255) & ~(size_t)255;     // keep 256B alignment
    return p;
  };
  // f32 scratch
  float* xin   = (float*)allocB((size_t)NM * 4);
  float* xz    = (float*)allocB((size_t)N_NODES * 2 * DI * 4);
  float* xc    = (float*)allocB((size_t)NM * 4);
  float* xdblg = (float*)allocB((size_t)N_NODES * 32 * 4);
  float* dpre  = (float*)allocB((size_t)NM * 4);
  float* dltg  = (float*)allocB((size_t)NM * 4);
  float* ga    = (float*)allocB((size_t)NM * 4);
  float* gAn   = (float*)allocB((size_t)DI * DS_Gc * 4);
  float* bAn   = (float*)allocB((size_t)DMD * DS_Bc * 4);
  float* ubuf  = (float*)allocB((size_t)L_GCN * NM * 4);
  float* tmp   = (float*)allocB((size_t)NM * 4);
  float* raw   = (float*)allocB((size_t)N_NODES * 64 * 4);
  float* dbuf  = (float*)allocB((size_t)L_GCN * NM * 4);
  float* Bbuf  = (float*)allocB((size_t)L_GCN * N_NODES * DS_Bc * 4);
  float* C7    = (float*)allocB((size_t)N_NODES * DS_Bc * 4);
  float* lpre  = (float*)allocB((size_t)NM * 4);
  float* gafin = (float*)allocB((size_t)NM * 4);
  // bf16 scratch (GEMM operands)
  __bf16* x_bf   = (__bf16*)allocB((size_t)N_NODES * F_IN * 2);
  __bf16* adj_bf = (__bf16*)allocB((size_t)N_NODES * N_NODES * 2);
  __bf16* xin_bf = (__bf16*)allocB((size_t)NM * 2);
  __bf16* lin1T  = (__bf16*)allocB((size_t)DMD * F_IN * 2);
  __bf16* ginw_b = (__bf16*)allocB((size_t)2 * DI * DMD * 2);
  __bf16* gxp_b  = (__bf16*)allocB((size_t)32 * DI * 2);
  __bf16* gdtT   = (__bf16*)allocB((size_t)DI * 32 * 2);
  __bf16* gout_b = (__bf16*)allocB((size_t)DMD * DI * 2);
  __bf16* xc_bf  = (__bf16*)allocB((size_t)NM * 2);
  __bf16* dt32g  = (__bf16*)allocB((size_t)N_NODES * 32 * 2);
  __bf16* yz_bf  = (__bf16*)allocB((size_t)NM * 2);
  __bf16* ubf    = (__bf16*)allocB((size_t)L_GCN * NM * 2);
  __bf16* featT  = (__bf16*)allocB((size_t)NM * 2);
  __bf16* bxT    = (__bf16*)allocB((size_t)64 * DMD * 2);
  __bf16* bdtT   = (__bf16*)allocB((size_t)DMD * 32 * 2);
  __bf16* boT    = (__bf16*)allocB((size_t)DMD * DMD * 2);
  __bf16* dt32l  = (__bf16*)allocB((size_t)N_NODES * 32 * 2);
  __bf16* yv_bf  = (__bf16*)allocB((size_t)NM * 2);

  auto gemm = [&](const __bf16* A, const __bf16* BT, float* C,
                  int M, int N, int K, int beta) {
    int tiles = (M / 16) * (N / 32);
    k_gemm<<<dim3((unsigned)((tiles + 7) / 8)), dim3(256), 0, stream>>>(A, BT, C, M, N, K, beta);
  };

  // ---- one-time conversions / weight preps ----
  k_cvt_bf<<<ew((size_t)N_NODES * F_IN), 256, 0, stream>>>(x_bf, x, N_NODES * F_IN);
  k_cvt_bf<<<ew((size_t)N_NODES * N_NODES), 256, 0, stream>>>(adj_bf, adj, N_NODES * N_NODES);
  k_transpose_bf<<<ew((size_t)DMD * F_IN), 256, 0, stream>>>(lin1T, lin1_w, F_IN, DMD, F_IN, DMD);
  k_cvt_bf<<<ew((size_t)2 * DI * DMD), 256, 0, stream>>>(ginw_b, g_in_w, 2 * DI * DMD);
  k_cvt_bf<<<ew((size_t)32 * DI), 256, 0, stream>>>(gxp_b, g_xproj, 32 * DI);
  k_padcopy_bf<<<ew((size_t)DI * 32), 256, 0, stream>>>(gdtT, g_dtproj, DI, DTR_Gc, 32);
  k_cvt_bf<<<ew((size_t)DMD * DI), 256, 0, stream>>>(gout_b, g_out_w, DMD * DI);
  k_transpose_bf<<<ew((size_t)64 * DMD), 256, 0, stream>>>(bxT, b_xproj, DMD, 48, DMD, 64);
  k_transpose_bf<<<ew((size_t)DMD * 32), 256, 0, stream>>>(bdtT, b_dtproj, DTR_Bc, DMD, 32, DMD);
  k_transpose_bf<<<ew((size_t)DMD * DMD), 256, 0, stream>>>(boT, b_out_w, DMD, DMD, DMD, DMD);
  k_neg_exp<<<ew((size_t)DI * DS_Gc), 256, 0, stream>>>(gAn, g_A_log, DI * DS_Gc);
  k_neg_exp<<<ew((size_t)DMD * DS_Bc), 256, 0, stream>>>(bAn, b_A_log, DMD * DS_Bc);

  // ---- x_input = x @ lin1_w ----
  gemm(x_bf, lin1T, xin, N_NODES, DMD, F_IN, 0);
  k_cvt_bf<<<ew((size_t)NM), 256, 0, stream>>>(xin_bf, xin, NM);

  // ---- global bidirectional Mamba ----
  gemm(xin_bf, ginw_b, xz, N_NODES, 2 * DI, DMD, 0);    // in-proj (weight already B^T)
  for (int dir = 0; dir < 2; ++dir) {
    k_conv_silu<<<ew((size_t)NM), 256, 0, stream>>>(xz, g_conv_w, g_conv_b, xc, xc_bf, dir);
    gemm(xc_bf, gxp_b, xdblg, N_NODES, 32, DI, 0);
    k_split_g<<<ew((size_t)N_NODES * 32), 256, 0, stream>>>(dt32g, xdblg);
    gemm(dt32g, gdtT, dpre, N_NODES, DI, 32, 0);
    k_softplus<<<ew((size_t)NM), 256, 0, stream>>>(dltg, dpre, g_dtb, NM, DI, 1);
    k_scan_global<<<dim3(1), dim3(256), 0, stream>>>(dltg, xc, xdblg, gAn, g_D, xz, yz_bf, dir);
    gemm(yz_bf, gout_b, ga, N_NODES, DMD, DI, dir);     // dir=1 accumulates
  }
  k_gafinal<<<ew((size_t)NM), 256, 0, stream>>>(gafin, ga, xin, NM);

  // ---- local GCN-Mamba path ----
  k_bn_relu<<<ew((size_t)NM), 256, 0, stream>>>(ubuf, ubf, xin,
                                                bn1_g, bn1_b, bn1_m, bn1_v, NM, DMD);
  for (int i = 1; i < L_GCN; ++i) {
    k_transpose_bf<<<ew((size_t)NM), 256, 0, stream>>>(featT, ubuf + (size_t)(i - 1) * NM,
                                                       N_NODES, DMD, N_NODES, DMD);
    gemm(adj_bf, featT, tmp, N_NODES, DMD, N_NODES, 0);
    k_blend<<<ew((size_t)NM), 256, 0, stream>>>(ubuf + (size_t)i * NM, ubf + (size_t)i * NM,
                                                tmp, ubuf, NM, BLKA);
  }
  for (int l = 0; l < L_GCN; ++l) {
    gemm(ubf + (size_t)l * NM, bxT, raw, N_NODES, 64, DMD, 0);
    k_split_b<<<ew((size_t)N_NODES * 64), 256, 0, stream>>>(
        dt32l, Bbuf + (size_t)l * N_NODES * DS_Bc, C7, raw);
    gemm(dt32l, bdtT, dpre, N_NODES, DMD, 32, 0);
    k_softplus<<<ew((size_t)NM), 256, 0, stream>>>(dbuf + (size_t)l * NM, dpre,
                                                   nullptr, NM, DMD, 0);
  }
  k_scan_block<<<ew((size_t)NM), 256, 0, stream>>>(yv_bf, ubuf, dbuf, Bbuf, C7, bAn, b_D);
  gemm(yv_bf, boT, lpre, N_NODES, DMD, DMD, 0);

  // ---- combine + outputs ----
  k_final<<<ew((size_t)NM), 256, 0, stream>>>((float*)d_out, lpre, xin, gafin,
                                              bn2_g, bn2_b, bn2_m, bn2_v);
}
